// WarmStartSinkhornLinear_73031623901730
// MI455X (gfx1250) — matrix operand order repaired and checked by reading, not compiled, not run on hardware
//
#include <hip/hip_runtime.h>

#define S 4096
#define MTOT 8192
#define TEMP_INV 10.0f
#define NEG_BIG (-3.0e38f)

typedef __attribute__((ext_vector_type(16))) __bf16 bf16x16;
typedef __attribute__((ext_vector_type(8)))  __bf16 bf16x8;
typedef __attribute__((ext_vector_type(8)))  float  f32x8;

__device__ __forceinline__ unsigned short f2bf(float f) {
    unsigned int u = __float_as_uint(f);
    u += 0x7FFFu + ((u >> 16) & 1u);   // round-to-nearest-even
    return (unsigned short)(u >> 16);
}

// ---------------------------------------------------------------- x -> bf16
__global__ __launch_bounds__(256) void convert_x_kernel(
    const float* __restrict__ in, unsigned short* __restrict__ outp) {
    size_t i = ((size_t)blockIdx.x * 256 + threadIdx.x) * 4;
    float4 v = *(const float4*)(in + i);
    uint2 pk;
    pk.x = (unsigned)f2bf(v.x) | ((unsigned)f2bf(v.y) << 16);
    pk.y = (unsigned)f2bf(v.z) | ((unsigned)f2bf(v.w) << 16);
    *(uint2*)(outp + i) = pk;
}

// ---------------------------------------------------------------- c init
__global__ __launch_bounds__(256) void init_c_kernel(
    const float* __restrict__ c_prev, float* __restrict__ cv) {
    int j = blockIdx.x * 256 + threadIdx.x;
    cv[j] = c_prev[j];
}

// ------------------------------------------------- r[i] = -lse_j(Q[i,j]+c[j])
// one wave32 per row, online logsumexp, shuffle tree reduction
__global__ __launch_bounds__(256) void row_lse_kernel(
    const float* __restrict__ W, const float* __restrict__ cv,
    float* __restrict__ rv) {
    const int row  = blockIdx.x * 8 + (threadIdx.x >> 5);
    const int lane = threadIdx.x & 31;
    const float* wr = W + (size_t)row * S;
    float m = NEG_BIG, s = 0.0f;
    for (int j = lane; j < S; j += 32) {
        float v = wr[j] * TEMP_INV + cv[j];
        if (v <= m) {
            s += __expf(v - m);
        } else {
            s = s * __expf(m - v) + 1.0f;
            m = v;
        }
    }
    #pragma unroll
    for (int off = 16; off > 0; off >>= 1) {
        float mo = __shfl_xor(m, off, 32);
        float so = __shfl_xor(s, off, 32);
        float mn = fmaxf(m, mo);
        s = s * __expf(m - mn) + so * __expf(mo - mn);
        m = mn;
    }
    if (lane == 0) rv[row] = -(m + __logf(s));
}

// ---------------------- column LSE: partial over 128-row chunks (coalesced)
__global__ __launch_bounds__(256) void col_lse_part_kernel(
    const float* __restrict__ W, const float* __restrict__ rv,
    float* __restrict__ pm, float* __restrict__ ps) {
    const int j  = blockIdx.x * 256 + threadIdx.x;   // column
    const int ck = blockIdx.y;                       // 0..31
    const int i0 = ck * (S / 32);
    float m = NEG_BIG, s = 0.0f;
    for (int i = i0; i < i0 + (S / 32); ++i) {
        float v = W[(size_t)i * S + j] * TEMP_INV + rv[i];
        if (v <= m) {
            s += __expf(v - m);
        } else {
            s = s * __expf(m - v) + 1.0f;
            m = v;
        }
    }
    pm[(size_t)ck * S + j] = m;
    ps[(size_t)ck * S + j] = s;
}

__global__ __launch_bounds__(256) void col_lse_comb_kernel(
    const float* __restrict__ pm, const float* __restrict__ ps,
    float* __restrict__ cv) {
    const int j = blockIdx.x * 256 + threadIdx.x;
    float m = NEG_BIG;
    #pragma unroll 4
    for (int k = 0; k < 32; ++k) m = fmaxf(m, pm[(size_t)k * S + j]);
    float s = 0.0f;
    #pragma unroll 4
    for (int k = 0; k < 32; ++k)
        s += ps[(size_t)k * S + j] * __expf(pm[(size_t)k * S + j] - m);
    cv[j] = -(m + __logf(s));
}

// ---------------------------------------- P[o,d] = bf16(exp(Q + r[o] + c[d]))
__global__ __launch_bounds__(256) void build_P_kernel(
    const float* __restrict__ W, const float* __restrict__ rv,
    const float* __restrict__ cv, unsigned short* __restrict__ P) {
    size_t i = ((size_t)blockIdx.x * 256 + threadIdx.x) * 4;
    int o = (int)(i >> 12);
    int d = (int)(i & (size_t)(S - 1));
    float4 w = *(const float4*)(W + i);
    float ro = rv[o];
    float e0 = __expf(w.x * TEMP_INV + ro + cv[d + 0]);
    float e1 = __expf(w.y * TEMP_INV + ro + cv[d + 1]);
    float e2 = __expf(w.z * TEMP_INV + ro + cv[d + 2]);
    float e3 = __expf(w.w * TEMP_INV + ro + cv[d + 3]);
    uint2 pk;
    pk.x = (unsigned)f2bf(e0) | ((unsigned)f2bf(e1) << 16);
    pk.y = (unsigned)f2bf(e2) | ((unsigned)f2bf(e3) << 16);
    *(uint2*)(P + i) = pk;
}

// ---------------------------------------------------------------- GEMM
// out[m,n] = sum_k A[m,k] * B[n,k]   (A = x bf16, B = P bf16, both K-major)
// Block tile 256(M) x 128(N), BK=32. 8 waves in 4x2 grid, 64x64 per wave:
// 16 WMMAs per 16 ds_load_b128 per wave per K-step.
#define BM 256
#define BN 128
#define BK 32
#define LSTRIDE 40   // halves per LDS row (32 data + 8 pad -> 80B rows)

__global__ __launch_bounds__(256) void gemm_bf16_kernel(
    const unsigned short* __restrict__ A,   // MTOT x S bf16
    const unsigned short* __restrict__ B,   // S x S bf16 (row = n, col = k)
    float* __restrict__ C) {                // MTOT x S f32
    __shared__ __align__(16) unsigned short lA[2][BM * LSTRIDE];  // 2 x 20480 B
    __shared__ __align__(16) unsigned short lB[2][BN * LSTRIDE];  // 2 x 10240 B

    const int tid   = threadIdx.x;
    const int lane  = tid & 31;
    const int wave  = tid >> 5;
    const int blkM  = blockIdx.y * BM;
    const int blkN  = blockIdx.x * BN;
    const int waveM = (wave >> 1) * 64;     // 0,64,128,192
    const int waveN = (wave & 1) * 64;      // 0,64

    // tile copy mapping: chunk c -> row = c>>2, half-col = (c&3)*8
    // A: 1024 chunks (4 per thread); B: 512 chunks (2 per thread)
    const int ar0 = (tid + 0)   >> 2, ah0 = ((tid + 0)   & 3) * 8;
    const int ar1 = (tid + 256) >> 2, ah1 = ((tid + 256) & 3) * 8;
    const int ar2 = (tid + 512) >> 2, ah2 = ((tid + 512) & 3) * 8;
    const int ar3 = (tid + 768) >> 2, ah3 = ((tid + 768) & 3) * 8;
    const int br0 = (tid + 0)   >> 2, bh0 = ((tid + 0)   & 3) * 8;
    const int br1 = (tid + 256) >> 2, bh1 = ((tid + 256) & 3) * 8;
    const size_t aOff0 = (size_t)(blkM + ar0) * S + ah0;
    const size_t aOff1 = (size_t)(blkM + ar1) * S + ah1;
    const size_t aOff2 = (size_t)(blkM + ar2) * S + ah2;
    const size_t aOff3 = (size_t)(blkM + ar3) * S + ah3;
    const size_t bOff0 = (size_t)(blkN + br0) * S + bh0;
    const size_t bOff1 = (size_t)(blkN + br1) * S + bh1;

    f32x8 acc[4][4] = {};
    uint4 ra0, ra1, ra2, ra3, rb0, rb1;

    // prologue: tile 0 -> regs -> LDS[0]
    ra0 = *(const uint4*)(A + aOff0);
    ra1 = *(const uint4*)(A + aOff1);
    ra2 = *(const uint4*)(A + aOff2);
    ra3 = *(const uint4*)(A + aOff3);
    rb0 = *(const uint4*)(B + bOff0);
    rb1 = *(const uint4*)(B + bOff1);
    *(uint4*)&lA[0][ar0 * LSTRIDE + ah0] = ra0;
    *(uint4*)&lA[0][ar1 * LSTRIDE + ah1] = ra1;
    *(uint4*)&lA[0][ar2 * LSTRIDE + ah2] = ra2;
    *(uint4*)&lA[0][ar3 * LSTRIDE + ah3] = ra3;
    *(uint4*)&lB[0][br0 * LSTRIDE + bh0] = rb0;
    *(uint4*)&lB[0][br1 * LSTRIDE + bh1] = rb1;
    __syncthreads();

    const int lr  = lane & 15;
    const int hiA = (lane >> 4) * 8;    // A frag: K{0-7,16-23} vs K{8-15,24-31}
    const int kbB = (lane >> 4) * 8;    // B frag: K{0-7,16-23} vs K{8-15,24-31}

    const int nK = S / BK;              // 128
    int buf = 0;
    for (int kt = 0; kt < nK; ++kt) {
        if (kt + 1 < nK) {
            const size_t kn = (size_t)(kt + 1) * BK;
            ra0 = *(const uint4*)(A + aOff0 + kn);
            ra1 = *(const uint4*)(A + aOff1 + kn);
            ra2 = *(const uint4*)(A + aOff2 + kn);
            ra3 = *(const uint4*)(A + aOff3 + kn);
            rb0 = *(const uint4*)(B + bOff0 + kn);
            rb1 = *(const uint4*)(B + bOff1 + kn);
            if (kt + 2 < nK) {
                __builtin_prefetch(A + aOff0 + kn + BK, 0, 1);
                __builtin_prefetch(B + bOff0 + kn + BK, 0, 1);
            }
        }

        bf16x16 bfr[4];
        #pragma unroll
        for (int fn = 0; fn < 4; ++fn) {
            const unsigned short* p = &lB[buf][(waveN + fn * 16 + lr) * LSTRIDE + kbB];
            bf16x8 x0 = *(const bf16x8*)(p);
            bf16x8 x1 = *(const bf16x8*)(p + 16);
            bfr[fn] = __builtin_shufflevector(x0, x1,
                0, 1, 2, 3, 4, 5, 6, 7, 8, 9, 10, 11, 12, 13, 14, 15);
        }

        #pragma unroll
        for (int fm = 0; fm < 4; ++fm) {
            const unsigned short* p = &lA[buf][(waveM + fm * 16 + lr) * LSTRIDE];
            bf16x8 x0 = *(const bf16x8*)(p + hiA);
            bf16x8 x1 = *(const bf16x8*)(p + hiA + 16);
            bf16x16 af = __builtin_shufflevector(x0, x1,
                0, 1, 2, 3, 4, 5, 6, 7, 8, 9, 10, 11, 12, 13, 14, 15);
            #pragma unroll
            for (int fn = 0; fn < 4; ++fn)
                acc[fm][fn] = __builtin_amdgcn_wmma_f32_16x16x32_bf16(
                    false, af, false, bfr[fn],
                    (short)0, acc[fm][fn], false, false);
        }

        if (kt + 1 < nK) {
            const int nb = buf ^ 1;
            *(uint4*)&lA[nb][ar0 * LSTRIDE + ah0] = ra0;
            *(uint4*)&lA[nb][ar1 * LSTRIDE + ah1] = ra1;
            *(uint4*)&lA[nb][ar2 * LSTRIDE + ah2] = ra2;
            *(uint4*)&lA[nb][ar3 * LSTRIDE + ah3] = ra3;
            *(uint4*)&lB[nb][br0 * LSTRIDE + bh0] = rb0;
            *(uint4*)&lB[nb][br1 * LSTRIDE + bh1] = rb1;
        }
        __syncthreads();
        buf ^= 1;
    }

    // epilogue: C/D layout: VGPR v, lane<16 -> (M=v, N=lane); lane>=16 -> (M=v+8, N=lane-16)
    #pragma unroll
    for (int fm = 0; fm < 4; ++fm) {
        const int mrow = blkM + waveM + fm * 16 + ((lane >> 4) << 3);
        #pragma unroll
        for (int fn = 0; fn < 4; ++fn) {
            const int ncol = blkN + waveN + fn * 16 + (lane & 15);
            float* o = C + (size_t)mrow * S + ncol;
            #pragma unroll
            for (int v = 0; v < 8; ++v)
                o[(size_t)v * S] = acc[fm][fn][v];
        }
    }
}

// ---------------------------------------------------------------- launch
extern "C" void kernel_launch(void* const* d_in, const int* in_sizes, int n_in,
                              void* d_out, int out_size, void* d_ws, size_t ws_size,
                              hipStream_t stream) {
    const float* x      = (const float*)d_in[0];   // [4,2048,4096]
    const float* W      = (const float*)d_in[1];   // [4096,4096]
    const float* c_prev = (const float*)d_in[3];   // [1,4096] (r_prev unused by algo)
    float* out = (float*)d_out;

    char* ws = (char*)d_ws;
    unsigned short* xbf = (unsigned short*)ws;                              // 64 MB
    unsigned short* Pbf = (unsigned short*)(ws + (size_t)MTOT * S * 2);     // 32 MB
    float* rv = (float*)(ws + (size_t)MTOT * S * 2 + (size_t)S * S * 2);
    float* cv = rv + S;
    float* pm = cv + S;              // 32 x S partial maxes
    float* ps = pm + 32 * S;         // 32 x S partial sums

    convert_x_kernel<<<(int)((size_t)MTOT * S / 1024), 256, 0, stream>>>(x, xbf);
    init_c_kernel<<<S / 256, 256, 0, stream>>>(c_prev, cv);

    for (int it = 0; it < 10; ++it) {
        row_lse_kernel<<<S / 8, 256, 0, stream>>>(W, cv, rv);
        col_lse_part_kernel<<<dim3(S / 256, 32), 256, 0, stream>>>(W, rv, pm, ps);
        col_lse_comb_kernel<<<S / 256, 256, 0, stream>>>(pm, ps, cv);
    }

    build_P_kernel<<<(int)((size_t)S * S / 1024), 256, 0, stream>>>(W, rv, cv, Pbf);
    gemm_bf16_kernel<<<dim3(S / BN, MTOT / BM), 256, 0, stream>>>(xbf, Pbf, out);
}
